// EnergyLayer_79886391705992
// MI455X (gfx1250) — matrix operand order repaired and checked by reading, not compiled, not run on hardware
//
#include <hip/hip_runtime.h>
#include <hip/hip_bf16.h>

typedef __attribute__((ext_vector_type(16))) _Float16 v16h;
typedef __attribute__((ext_vector_type(8)))  _Float16 v8h;
typedef __attribute__((ext_vector_type(4)))  _Float16 v4h;
typedef __attribute__((ext_vector_type(8)))  float    v8f;

#define NF    128   // NUM_IN == NUM_OUT
#define ROWS  128   // rows per block
#define PITCH 136   // LDS row pitch in halfs (stride 68 dwords -> conflict-free)
#define NSEG  256   // B

// largest i in [0,255] with off[i] <= p  (off has 257 entries, off[0]=0)
__device__ __forceinline__ int find_seg(const int* __restrict__ off, int p) {
    int lo = 0, hi = NSEG;
    while (hi - lo > 1) {
        int mid = (lo + hi) >> 1;
        if (off[mid] <= p) lo = mid; else hi = mid;
    }
    return lo;
}

__device__ __forceinline__ float fast_silu(float h) {
    // h * sigmoid(h); v_exp_f32 + v_rcp_f32, no IEEE divide
    return h * __builtin_amdgcn_rcpf(1.0f + __expf(-h));
}

__global__ void init_prefix(const int* __restrict__ sizes,
                            const int* __restrict__ psizes,
                            int* __restrict__ offD, int* __restrict__ offO) {
    if (threadIdx.x == 0) {
        int a = 0, b = 0;
        offD[0] = 0; offO[0] = 0;
        for (int i = 0; i < NSEG; ++i) {
            a += sizes[i];  offD[i + 1] = a;
            b += psizes[i]; offO[i + 1] = b;
        }
    }
}

__global__ void zero_acc(float* __restrict__ accD, float* __restrict__ accO) {
    int i = blockIdx.x * blockDim.x + threadIdx.x;   // 128 blocks * 256 = 32768
    accD[i] = 0.0f;
    accO[i] = 0.0f;
}

// X: [nrows, 128] f32 row-major.  W: [128 out, 128 in] f32 row-major.
// Computes silu(X @ W^T + bias) and segment-sums rows into acc[seg][128].
template<bool TWOSEG>
__global__ __launch_bounds__(256) void gemm_silu_seg(
    const float* __restrict__ X, const float* __restrict__ W,
    const float* __restrict__ bias, const int* __restrict__ off,
    float* __restrict__ acc, int nrows)
{
    __shared__ alignas(16) _Float16 sW[NF * PITCH];
    __shared__ alignas(16) _Float16 sX[ROWS * PITCH];
    __shared__ float segacc[2 * NF];
    __shared__ float sbias[NF];
    __shared__ int   srow_seg[ROWS];
    __shared__ int   sseg0, sboundary, shas2;

    const int t  = threadIdx.x;
    const int r0 = blockIdx.x * ROWS;

    // ---- stage W: f32 -> f16 into LDS (coalesced float4 loads) ----
    const float4* W4 = reinterpret_cast<const float4*>(W);
    #pragma unroll
    for (int i = 0; i < 16; ++i) {
        int idx = t + i * 256;                 // 4096 float4 total
        float4 v = W4[idx];
        int e = idx << 2, n = e >> 7, k = e & (NF - 1);
        v4h h = { (_Float16)v.x, (_Float16)v.y, (_Float16)v.z, (_Float16)v.w };
        *reinterpret_cast<v4h*>(&sW[n * PITCH + k]) = h;
    }
    // ---- stage X rows [r0, r0+128), zero-fill past nrows ----
    const float4* X4 = reinterpret_cast<const float4*>(X);
    #pragma unroll
    for (int i = 0; i < 16; ++i) {
        int idx = t + i * 256;
        int e = idx << 2, lr = e >> 7, k = e & (NF - 1);
        int p = r0 + lr;
        float4 v = make_float4(0.f, 0.f, 0.f, 0.f);
        if (p < nrows) v = X4[(size_t)p * (NF / 4) + (k >> 2)];
        v4h h = { (_Float16)v.x, (_Float16)v.y, (_Float16)v.z, (_Float16)v.w };
        *reinterpret_cast<v4h*>(&sX[lr * PITCH + k]) = h;
    }
    if (t < NF) sbias[t] = bias[t];
    if (TWOSEG) {
        segacc[t] = 0.0f;                      // 256 floats, one per thread
        if (t == 0) {
            int s0 = find_seg(off, r0);
            sseg0 = s0;
            int bnd = off[s0 + 1];
            sboundary = bnd;
            int lim = (r0 + ROWS < nrows) ? (r0 + ROWS) : nrows;
            shas2 = (bnd < lim) ? 1 : 0;
        }
    } else {
        if (t < ROWS) {
            int p = r0 + t;
            srow_seg[t] = (p < nrows) ? find_seg(off, p) : 0;
        }
    }
    __syncthreads();

    const int wv   = t >> 5;       // wave id 0..7 (wave32)
    const int lane = t & 31;
    const int l16  = lane & 15;
    const int hsel = lane >> 4;
    const int mrow = wv * 16 + l16;

    // ---- A fragments: 16-bit A 16x32 layout.
    // lane l, kstep s: K = s*32 + hsel*8 + {0..7} (v0-3) and +16 (v4-7)
    v16h afr[4];
    #pragma unroll
    for (int s = 0; s < 4; ++s) {
        const _Float16* pa = &sX[mrow * PITCH + s * 32 + hsel * 8];
        v8h lo = *reinterpret_cast<const v8h*>(pa);
        v8h hi = *reinterpret_cast<const v8h*>(pa + 16);
        afr[s] = __builtin_shufflevector(lo, hi, 0,1,2,3,4,5,6,7,8,9,10,11,12,13,14,15);
    }

    // ---- per-row epilogue metadata, hoisted out of the column-tile loop.
    // Branch-free epilogue: invalid (padded) rows contribute 0.0 to seg 0.
    float rmask[8];
    int   rseg[8];
    #pragma unroll
    for (int r = 0; r < 8; ++r) {
        int lr = wv * 16 + r + hsel * 8;
        int p  = r0 + lr;
        bool ok = (p < nrows);
        rmask[r] = ok ? 1.0f : 0.0f;
        if (TWOSEG) rseg[r] = (ok && p >= sboundary) ? 1 : 0;
        else        rseg[r] = ok ? srow_seg[lr] : 0;
    }

    #pragma unroll
    for (int n = 0; n < 8; ++n) {
        v8f c = {};
        #pragma unroll
        for (int s = 0; s < 4; ++s) {
            // B 32x16: lane l = column (W row n*16+l16); K = hsel*8 + {0..7}
            // (v0-3) and +16 (v4-7), matching the A layout.
            const _Float16* pb = &sW[(n * 16 + l16) * PITCH + s * 32 + hsel * 8];
            v8h lo = *reinterpret_cast<const v8h*>(pb);
            v8h hi = *reinterpret_cast<const v8h*>(pb + 16);
            v16h bfr = __builtin_shufflevector(lo, hi, 0,1,2,3,4,5,6,7,8,9,10,11,12,13,14,15);
            c = __builtin_amdgcn_wmma_f32_16x16x32_f16(
                    false, afr[s], false, bfr, (short)0, c, false, false);
        }
        // C/D layout: c[r] = element (M = r + hsel*8, N = l16) of the tile
        const int col = n * 16 + l16;
        const float bc = sbias[col];
        #pragma unroll
        for (int r = 0; r < 8; ++r) {
            float h = fast_silu(c[r] + bc) * rmask[r];
            if (TWOSEG) {
                atomicAdd(&segacc[rseg[r] * NF + col], h);   // ds_add_f32
            } else {
                atomicAdd(&acc[rseg[r] * NF + col], h);      // global_atomic_add_f32
            }
        }
    }

    if (TWOSEG) {
        __syncthreads();
        if (t < NF) {
            atomicAdd(&acc[sseg0 * NF + t], segacc[t]);
        } else if (shas2) {
            int col = t - NF;
            atomicAdd(&acc[(sseg0 + 1) * NF + col], segacc[NF + col]);
        }
    }
}

__global__ void finalize(const float* __restrict__ accD, const float* __restrict__ accO,
                         const int* __restrict__ sizes, const int* __restrict__ psizes,
                         const float* __restrict__ Wout, const float* __restrict__ bout,
                         float* __restrict__ out) {
    int b = threadIdx.x;                         // 256 threads
    float invd = 1.0f / (float)sizes[b];
    float invo = 1.0f / (float)psizes[b];
    float e = bout[0];
    #pragma unroll 8
    for (int f = 0; f < NF; ++f) {
        e += accD[b * NF + f] * invd * Wout[f];
        e += accO[b * NF + f] * invo * Wout[NF + f];
    }
    out[b] = e;
}

extern "C" void kernel_launch(void* const* d_in, const int* in_sizes, int n_in,
                              void* d_out, int out_size, void* d_ws, size_t ws_size,
                              hipStream_t stream) {
    const float* fii    = (const float*)d_in[0];
    const float* fij    = (const float*)d_in[1];
    const int*   sizes  = (const int*)  d_in[2];
    const int*   psizes = (const int*)  d_in[3];
    const float* Wd     = (const float*)d_in[4];
    const float* bd     = (const float*)d_in[5];
    const float* Wo     = (const float*)d_in[6];
    const float* bo     = (const float*)d_in[7];
    const float* Wout   = (const float*)d_in[8];
    const float* bout   = (const float*)d_in[9];
    float* out = (float*)d_out;

    const int N = in_sizes[0] / NF;   // 7646
    const int P = in_sizes[1] / NF;   // 230248

    float* accD = (float*)d_ws;                 // [256][128]
    float* accO = accD + NSEG * NF;             // [256][128]
    int*   offD = (int*)(accO + NSEG * NF);     // [257]
    int*   offO = offD + (NSEG + 1);            // [257]

    zero_acc<<<(NSEG * NF) / 256, 256, 0, stream>>>(accD, accO);
    init_prefix<<<1, 32, 0, stream>>>(sizes, psizes, offD, offO);

    gemm_silu_seg<false><<<(N + ROWS - 1) / ROWS, 256, 0, stream>>>(
        fii, Wd, bd, offD, accD, N);
    gemm_silu_seg<true><<<(P + ROWS - 1) / ROWS, 256, 0, stream>>>(
        fij, Wo, bo, offO, accO, P);

    finalize<<<1, 256, 0, stream>>>(accD, accO, sizes, psizes, Wout, bout, out);
}
